// BackboneEncoderFirstStage_Mamba_38878043963456
// MI455X (gfx1250) — matrix operand order repaired and checked by reading, not compiled, not run on hardware
//
#include <hip/hip_runtime.h>

// ---------------------------------------------------------------------------
// Types for CDNA5 WMMA (wave32, 16x16x32 f16 -> f32)
// ---------------------------------------------------------------------------
typedef __attribute__((ext_vector_type(16))) _Float16 v16h;
typedef __attribute__((ext_vector_type(8)))  _Float16 half8;
typedef __attribute__((ext_vector_type(8)))  float    v8f;

union V16 { v16h v; half8 h[2]; };

__device__ inline half8 h8_zero() {
  half8 z;
#pragma unroll
  for (int i = 0; i < 8; ++i) z[i] = (_Float16)0.f;
  return z;
}

// ---------------------------------------------------------------------------
// Tensor Data Mover (async global->LDS DMA) support, gated per toolchain:
// ROCm 7.2 (clang-22): 5-arg builtin; amdgpu-toolchain (clang-23+): 6-arg.
// ---------------------------------------------------------------------------
#if defined(__has_builtin)
#  if __has_builtin(__builtin_amdgcn_tensor_load_to_lds) && \
      __has_builtin(__builtin_amdgcn_s_wait_tensorcnt) && \
      defined(__clang_major__) && (__clang_major__ >= 22)
#    define USE_TDM 1
#  endif
#endif
#ifndef USE_TDM
#  define USE_TDM 0
#endif

typedef __attribute__((ext_vector_type(4))) unsigned int u32x4;
typedef __attribute__((ext_vector_type(4))) int          i32x4v;
typedef __attribute__((ext_vector_type(8))) int          i32x8v;

#if USE_TDM
// 2-D tile load: rows x 32 f16 elements, row stride = strideElems, into LDS
// at byte offset ldsOff. D# per CDNA5 ISA ch.8: group0 = control/lds/global
// address/type=2; group1 = data_size(2B), tensor dims (OOB rows read zero),
// tile 32 x 128, dim0 stride; groups 2/3 zero (<=2D tensor).
__device__ inline void tdm_load_tile_2d(unsigned ldsOff, const _Float16* gsrc,
                                        int rowsAvail, int kAvail, int strideElems)
{
  unsigned long long ga = (unsigned long long)(size_t)gsrc;
  u32x4 g0;
  g0[0] = 1u;                                              // count=1, user D#
  g0[1] = ldsOff;                                          // lds_addr (bytes)
  g0[2] = (unsigned)(ga & 0xffffffffu);                    // global_addr[31:0]
  g0[3] = (unsigned)((ga >> 32) & 0x01ffffffu) | (2u << 30); // [56:32] | type=2
  unsigned td0 = (unsigned)kAvail;                         // tensor_dim0
  unsigned td1 = (unsigned)rowsAvail;                      // tensor_dim1
  unsigned long long st0 = (unsigned long long)(unsigned)strideElems;
  i32x8v g1;
  g1[0] = (int)(1u << 16);                                 // data_size = 2B
  g1[1] = (int)((td0 & 0xffffu) << 16);                    // tensor_dim0 lo
  g1[2] = (int)((td0 >> 16) | ((td1 & 0xffffu) << 16));    // td0 hi | td1 lo
  g1[3] = (int)((td1 >> 16) | (32u << 16));                // td1 hi | tile_dim0=32
  g1[4] = (int)128;                                        // tile_dim1=128
  g1[5] = (int)(st0 & 0xffffffffu);                        // dim0_stride lo
  g1[6] = (int)((st0 >> 32) & 0xffffu);                    // dim0_stride hi
  g1[7] = 0;                                               // dim1_stride (unused 2D)
  i32x4v z4; z4[0] = 0; z4[1] = 0; z4[2] = 0; z4[3] = 0;
#if __clang_major__ >= 23
  i32x8v z8; z8[0]=0; z8[1]=0; z8[2]=0; z8[3]=0; z8[4]=0; z8[5]=0; z8[6]=0; z8[7]=0;
  __builtin_amdgcn_tensor_load_to_lds(g0, g1, z4, z4, z8, 0);
#else
  __builtin_amdgcn_tensor_load_to_lds(g0, g1, z4, z4, 0);
#endif
}
#endif

// ---------------------------------------------------------------------------
// Generic implicit-GEMM conv / GEMM kernel using v_wmma_f32_16x16x32_f16.
//
//   Out[pix][co] = epilogue( sum_k A[co][k] * B[k][pix] )
//   A = repacked weights [Cout][K] f16, kk = (r*S+s)*Cin + ci
//   B = NHWC f16 activations, implicit im2col gathered into LDS
//
// Block tile: 64 pixels x 128 Cout. 8 waves, each a 32x32 tile (2x2 register
// blocking -> 4 WMMAs / K-step). Per 32-K step the block stages:
//   A slice 128x32 f16 (8 KB): TDM async DMA (tensor_load_to_lds) when
//     available, else cooperative global->LDS copy.
//   B slice 64x32 f16 (4 KB): cooperative implicit-im2col gather (division-
//     free incremental (r,s,ci) walk; Cin>=32 guaranteed).
// Double-buffered LDS (24 KB), one barrier per step. Guards are wave-uniform
// so EXEC is all-ones at every WMMA; no thread exits early (barrier safety).
// Host guarantees K%32==0, Cin%16==0 when R*S>1, M%16==0, Cout%16==0.
// ---------------------------------------------------------------------------
__global__ void __launch_bounds__(256) k_conv_wmma(
    const _Float16* __restrict__ In, const _Float16* __restrict__ W,
    float* __restrict__ Out32, _Float16* __restrict__ Out16,
    int Bn, int Hin, int Win, int Cin, int ldin,
    int Hout, int Wout, int Cout, int ldout,
    int K, int R, int S, int pad, int stride,
    const float* __restrict__ scale, const float* __restrict__ bias,
    const float* __restrict__ alpha,
    const float* __restrict__ res, int resMode, int ldres)
{
  __shared__ _Float16 lA[2][128 * 32];
  __shared__ _Float16 lB[2][64 * 32];

  const int tid  = threadIdx.x;
  const int wave = tid >> 5;
  const int lane = tid & 31;
  const int M  = Bn * Hout * Wout;
  const int hw = Hout * Wout;

  const int pixBlock = blockIdx.x * 64;
  const int coBlock  = blockIdx.y * 128;
  const int pixWave  = pixBlock + (wave >> 2) * 32;  // 2 pixel super-tiles
  const int coWave   = coBlock + (wave & 3) * 32;    // 4 cout super-tiles

  const int m  = lane & 15;
  const int hi = lane >> 4;

  // wave-uniform sub-tile validity (tiles are 16-granular, M%16==0)
  bool vP[2], vC[2];
  vP[0] = pixWave < M;        vP[1] = (pixWave + 16) < M;
  vC[0] = coWave < Cout;      vC[1] = (coWave + 16) < Cout;

#if !USE_TDM
  // A: 128 rows x 4 chunks of 8 halfs = 512 segments, 2 per thread
  const int arow = tid >> 2;
  const int achk = (tid & 3) * 8;
#endif
  // B: 64 pixels x 4 chunks of 8 halfs = 256 segments, 1 per thread
  const int fpix = tid >> 2;            // 0..63
  const int fchk = (tid & 3) * 8;       // 0,8,16,24
  const int gpix = pixBlock + fpix;
  const bool fvalid = gpix < M;
  int fb = 0, iyBase = 0, ixBase = 0;
  if (fvalid) {
    fb = gpix / hw;
    int rem = gpix - fb * hw;
    int foy = rem / Wout;
    int fox = rem - foy * Wout;
    iyBase = foy * stride - pad;
    ixBase = fox * stride - pad;
  }
  // division-free im2col walk: k = kb + fchk, fchk < 32 <= Cin  =>  initial
  // state ci=fchk, r=s=0; k advances by 32/step, at most one rs-wrap/step.
  int ciB = fchk, rB = 0, sB = 0;

  v8f zero = {};
  v8f acc[2][2];
  acc[0][0] = zero; acc[0][1] = zero; acc[1][0] = zero; acc[1][1] = zero;

  const int lp = (wave >> 2) * 32;      // local pixel base in lB
  const int lc = (wave & 3) * 32;       // local cout base in lA

  int st = 0;
#pragma unroll 2
  for (int kb = 0; kb < K; kb += 32, st ^= 1) {
#if USE_TDM
    // ---- stage A slice via Tensor Data Mover (async DMA, EXEC-independent;
    //      OOB rows beyond (Cout-coBlock) read as zero) ----
    if (tid == 0) {
      tdm_load_tile_2d((unsigned)(size_t)&lA[st][0],
                       W + (size_t)coBlock * K + kb,
                       Cout - coBlock, K - kb, K);
    }
#else
    // ---- stage A slice cooperatively (weights stream) ----
    {
      const int gr0 = coBlock + arow;
      const int gr1 = gr0 + 64;
      half8 va0 = h8_zero(), va1 = h8_zero();
      if (gr0 < Cout) {
        const _Float16* p = W + (size_t)gr0 * K + kb + achk;
        __builtin_prefetch(p + 64, 0, 1);       // global_prefetch_b8
        va0 = *(const half8*)p;
      }
      if (gr1 < Cout) va1 = *(const half8*)(W + (size_t)gr1 * K + kb + achk);
      *(half8*)&lA[st][arow * 32 + achk]        = va0;
      *(half8*)&lA[st][(arow + 64) * 32 + achk] = va1;
    }
#endif
    // ---- stage B slice (implicit im2col gather, once per block) ----
    {
      half8 vb = h8_zero();
      if (fvalid) {
        const int iy = iyBase + rB;
        const int ix = ixBase + sB;
        if (iy >= 0 && iy < Hin && ix >= 0 && ix < Win)
          vb = *(const half8*)(In +
                ((size_t)(fb * Hin + iy) * Win + ix) * (size_t)ldin + ciB);
      }
      *(half8*)&lB[st][fpix * 32 + fchk] = vb;
    }
    // advance (ci, s, r) by 32 K (at most one wrap since Cin >= 32)
    ciB += 32;
    if (ciB >= Cin) { ciB -= Cin; ++sB; if (sB == S) { sB = 0; ++rB; } }

#if USE_TDM
    if (tid == 0) __builtin_amdgcn_s_wait_tensorcnt(0);   // drain TDM
#endif
    __syncthreads();

    // ---- load fragments from LDS (ds_load_b128) ----
    V16 a0, a1, b0, b1;
    a0.h[0] = *(const half8*)&lA[st][(lc + m) * 32 + hi * 8];
    a0.h[1] = *(const half8*)&lA[st][(lc + m) * 32 + 16 + hi * 8];
    a1.h[0] = *(const half8*)&lA[st][(lc + 16 + m) * 32 + hi * 8];
    a1.h[1] = *(const half8*)&lA[st][(lc + 16 + m) * 32 + 16 + hi * 8];
    b0.h[0] = *(const half8*)&lB[st][(lp + m) * 32 + hi * 8];
    b0.h[1] = *(const half8*)&lB[st][(lp + m) * 32 + 16 + hi * 8];
    b1.h[0] = *(const half8*)&lB[st][(lp + 16 + m) * 32 + hi * 8];
    b1.h[1] = *(const half8*)&lB[st][(lp + 16 + m) * 32 + 16 + hi * 8];

    // ---- 2x2 register-blocked WMMA (guards are wave-uniform) ----
    if (vC[0] && vP[0])
      acc[0][0] = __builtin_amdgcn_wmma_f32_16x16x32_f16(
          false, a0.v, false, b0.v, (short)0, acc[0][0], false, false);
    if (vC[0] && vP[1])
      acc[0][1] = __builtin_amdgcn_wmma_f32_16x16x32_f16(
          false, a0.v, false, b1.v, (short)0, acc[0][1], false, false);
    if (vC[1] && vP[0])
      acc[1][0] = __builtin_amdgcn_wmma_f32_16x16x32_f16(
          false, a1.v, false, b0.v, (short)0, acc[1][0], false, false);
    if (vC[1] && vP[1])
      acc[1][1] = __builtin_amdgcn_wmma_f32_16x16x32_f16(
          false, a1.v, false, b1.v, (short)0, acc[1][1], false, false);
  }

  // ---- epilogue + store (D layout: lane&15 = pixel, hi*8+v = cout) ----
#pragma unroll
  for (int j = 0; j < 2; ++j) {
    if (!vC[j]) continue;
#pragma unroll
    for (int i = 0; i < 2; ++i) {
      if (!vP[i]) continue;
      const int pixD = pixWave + i * 16 + m;
      const int bD   = pixD / hw;
      const int remD = pixD - bD * hw;
      const int oyD  = remD / Wout;
      const int oxD  = remD - oyD * Wout;
      const int co0  = coWave + j * 16 + hi * 8;

      float outv[8];
#pragma unroll
      for (int v = 0; v < 8; ++v) {
        const int co = co0 + v;
        float x = acc[j][i][v];
        if (scale)      x = x * scale[co] + (bias ? bias[co] : 0.f);
        else if (bias)  x += bias[co];
        if (alpha)      x = x >= 0.f ? x : alpha[co] * x;
        if (resMode == 1)
          x += res[(size_t)pixD * ldres + co];
        else if (resMode == 2)
          x += res[((size_t)(bD * Hin + oyD * stride) * Win + oxD * stride)
                       * (size_t)ldres + co];
        outv[v] = x;
      }
      if (Out32) {
        float* o = Out32 + (size_t)pixD * ldout + co0;
#pragma unroll
        for (int v = 0; v < 8; ++v) o[v] = outv[v];
      }
      if (Out16) {
        _Float16* o = Out16 + (size_t)pixD * ldout + co0;
#pragma unroll
        for (int v = 0; v < 8; ++v) o[v] = (_Float16)outv[v];
      }
    }
  }
}

// ---------------------------------------------------------------------------
// BN parameter folding: s = g/sqrt(v+eps), b' = b - m*s
// ---------------------------------------------------------------------------
__global__ void k_bnfold(const float* __restrict__ g, const float* __restrict__ b,
                         const float* __restrict__ m, const float* __restrict__ v,
                         float* __restrict__ s, float* __restrict__ bo, int C)
{
  int c = blockIdx.x * 256 + threadIdx.x;
  if (c >= C) return;
  float sc = g[c] * rsqrtf(v[c] + 1e-5f);
  s[c]  = sc;
  bo[c] = b[c] - m[c] * sc;
}

// ---------------------------------------------------------------------------
// Per-channel affine + cast f32 -> f16 (NHWC, channel = idx % C; every C we
// use is a power of two -> mask fast path)
// ---------------------------------------------------------------------------
__global__ void k_affine16(const float* __restrict__ src, _Float16* __restrict__ dst,
                           long long n, int C, const float* __restrict__ s,
                           const float* __restrict__ b)
{
  long long i = (long long)blockIdx.x * 256 + threadIdx.x;
  if (i >= n) return;
  int c = ((C & (C - 1)) == 0) ? (int)(i & (C - 1)) : (int)(i % C);
  float v = src[i];
  if (s) v = v * s[c] + (b ? b[c] : 0.f);
  dst[i] = (_Float16)v;
}

// ---------------------------------------------------------------------------
// Weight repack: OIHW f32 -> [O][K] f16 with kk = (r*S+s)*I + i
// ---------------------------------------------------------------------------
__global__ void k_repack(const float* __restrict__ w, _Float16* __restrict__ dst,
                         int O, int I, int R, int S)
{
  long long K = (long long)I * R * S;
  long long idx = (long long)blockIdx.x * 256 + threadIdx.x;
  if (idx >= (long long)O * K) return;
  int o  = (int)(idx / K);
  int kk = (int)(idx % K);
  int rs = kk / I;
  int i2 = kk - rs * I;
  int r  = rs / S;
  int s  = rs - r * S;
  dst[idx] = (_Float16)w[(((long long)o * I + i2) * R + r) * S + s];
}

// ---------------------------------------------------------------------------
// Input conv: NCHW f32 (16,3,256,256) -> NHWC f32 (16,256,256,64), 3x3 pad1,
// fused folded-BN + PReLU. Only 27 MACs/output; direct kernel is fine.
// ---------------------------------------------------------------------------
__global__ void k_input_conv(const float* __restrict__ x, const float* __restrict__ w,
                             const float* __restrict__ s, const float* __restrict__ bb,
                             const float* __restrict__ al, float* __restrict__ out)
{
  long long idx = (long long)blockIdx.x * 256 + threadIdx.x;
  const long long total = (long long)16 * 256 * 256 * 64;
  if (idx >= total) return;
  int co = (int)(idx & 63);
  long long pix = idx >> 6;
  int b   = (int)(pix >> 16);
  int rem = (int)(pix & 65535);
  int y   = rem >> 8;
  int xp  = rem & 255;
  float a = 0.f;
  for (int ci = 0; ci < 3; ++ci)
    for (int r = 0; r < 3; ++r) {
      int iy = y + r - 1;
      if (iy < 0 || iy >= 256) continue;
      for (int sp = 0; sp < 3; ++sp) {
        int ix = xp + sp - 1;
        if (ix < 0 || ix >= 256) continue;
        a += x[(((long long)b * 3 + ci) * 256 + iy) * 256 + ix] *
             w[((co * 3 + ci) * 3 + r) * 3 + sp];
      }
    }
  a = a * s[co] + bb[co];
  a = a >= 0.f ? a : al[co] * a;
  out[idx] = a;
}

// ---------------------------------------------------------------------------
// RMSNorm + residual accumulate. One block per token (512 features),
// LDS tree reduction for sum of squares. Writes res_out f32 and hn f16.
// ---------------------------------------------------------------------------
__global__ void __launch_bounds__(256) k_rmsnorm_res(
    const float* __restrict__ h, const float* __restrict__ resIn,
    float* __restrict__ resOut, _Float16* __restrict__ hn,
    const float* __restrict__ w)
{
  __shared__ float red[256];
  const int t = blockIdx.x;
  const size_t base = (size_t)t * 512;
  const int c0 = threadIdx.x, c1 = threadIdx.x + 256;
  float r0 = h[base + c0], r1 = h[base + c1];
  if (resIn) { r0 += resIn[base + c0]; r1 += resIn[base + c1]; }
  resOut[base + c0] = r0;
  resOut[base + c1] = r1;
  red[threadIdx.x] = r0 * r0 + r1 * r1;
  __syncthreads();
  for (int sft = 128; sft > 0; sft >>= 1) {
    if (threadIdx.x < sft) red[threadIdx.x] += red[threadIdx.x + sft];
    __syncthreads();
  }
  const float inv = rsqrtf(red[0] * (1.f / 512.f) + 1e-5f);
  hn[base + c0] = (_Float16)(r0 * inv * w[c0]);
  hn[base + c1] = (_Float16)(r1 * inv * w[c1]);
}

// ---------------------------------------------------------------------------
// Depthwise causal conv1d (D_CONV=4) + SiLU for one Mamba direction.
// Reads x_ slice of xz (stride 2048) at physical token order, writes xl at
// LOGICAL sequence order (so downstream GEMMs stay contiguous).
// ---------------------------------------------------------------------------
__global__ void k_conv1d_silu(const float* __restrict__ xz,
                              const float* __restrict__ cw, const float* __restrict__ cb,
                              float* __restrict__ xl32, _Float16* __restrict__ xl16,
                              int rev)
{
  int idx = blockIdx.x * 256 + threadIdx.x;
  if (idx >= 16 * 64 * 1024) return;
  int d = idx & 1023;
  int t = (idx >> 10) & 63;
  int b = idx >> 16;
  float a = cb[d];
#pragma unroll
  for (int j = 0; j < 4; ++j) {
    int ts = t - 3 + j;
    if (ts < 0) continue;
    int phys = b * 64 + (rev ? 63 - ts : ts);
    a += xz[(size_t)phys * 2048 + d] * cw[d * 4 + j];
  }
  float sv = a / (1.f + __expf(-a));          // SiLU
  size_t o = (size_t)(b * 64 + t) * 1024 + d;
  xl32[o] = sv;
  xl16[o] = (_Float16)sv;
}

// ---------------------------------------------------------------------------
// Selective scan: one thread per (batch, inner-channel); 16-entry state kept
// in registers, 64 sequential steps. Heavy v_exp_f32 (TRANS pipe) + VALU.
// Fuses softplus(dt), dA, dBx, C-reduction, skip (D), SiLU(z) gate, and the
// 0.5 * (fwd + bwd) combination (accum: 0 = store, 1 = add).
// ---------------------------------------------------------------------------
__global__ void k_scan(const float* __restrict__ dtp, const float* __restrict__ dtb,
                       const float* __restrict__ alog, const float* __restrict__ xdbl,
                       const float* __restrict__ xl, const float* __restrict__ Dp,
                       const float* __restrict__ xz, float* __restrict__ yout,
                       int rev, int accum)
{
  int tid = blockIdx.x * 256 + threadIdx.x;
  if (tid >= 16 * 1024) return;
  int b = tid >> 10;
  int d = tid & 1023;
  float A[16], h[16];
#pragma unroll
  for (int n = 0; n < 16; ++n) {
    A[n] = -__expf(alog[d * 16 + n]);
    h[n] = 0.f;
  }
  const float Dd = Dp[d];
  const float db = dtb[d];
  for (int t = 0; t < 64; ++t) {
    const int tok = b * 64 + t;
    float dtv = dtp[(size_t)tok * 1024 + d] + db;
    float delta = dtv > 20.f ? dtv : log1pf(__expf(dtv));   // softplus
    float x = xl[(size_t)tok * 1024 + d];
    const float* bc = xdbl + (size_t)tok * 64;
    float y = 0.f;
#pragma unroll
    for (int n = 0; n < 16; ++n) {
      float dA = __expf(delta * A[n]);
      h[n] = dA * h[n] + delta * bc[32 + n] * x;
      y += h[n] * bc[48 + n];
    }
    y += x * Dd;
    const int tp   = rev ? 63 - t : t;
    const int ptok = b * 64 + tp;
    float z = xz[(size_t)ptok * 2048 + 1024 + d];
    float g = z / (1.f + __expf(-z));
    float val = 0.5f * y * g;
    size_t oi = (size_t)ptok * 1024 + d;
    if (accum) yout[oi] += val; else yout[oi] = val;
  }
}

// ---------------------------------------------------------------------------
// Head pre-reduction: BN over token dim (64 "channels"), then mean over
// groups of 8 along the 512 feature dim -> feat f16 [16][4096].
// ---------------------------------------------------------------------------
__global__ void k_head_pre(const float* __restrict__ x1, const float* __restrict__ s,
                           const float* __restrict__ bb, _Float16* __restrict__ feat)
{
  int idx = blockIdx.x * 256 + threadIdx.x;
  if (idx >= 16 * 4096) return;
  int g   = idx & 63;
  int tok = (idx >> 6) & 63;
  int b   = idx >> 12;
  float sc = s[tok], bo = bb[tok];
  float a = 0.f;
#pragma unroll
  for (int j = 0; j < 8; ++j)
    a += x1[(size_t)(b * 64 + tok) * 512 + g * 8 + j] * sc + bo;
  feat[(size_t)b * 4096 + tok * 64 + g] = (_Float16)(a * 0.125f);
}

// ===========================================================================
// Host orchestration
// ===========================================================================
struct BNp  { const float *g, *b, *m, *v; };
struct Block { BNp bn1; const float *conv1, *prelu, *conv2; BNp bn2;
               const float* scw; BNp scbn; bool has_sc; };
struct DirP { const float *cw, *cb, *xw, *dw, *db, *alog, *D; };
struct Layer { const float *norm, *inw, *outw; DirP dir[2]; };
struct Head { BNp bn; const float *lw, *lb; };

static const int CFG[21][3] = {
  {64,64,2},{64,64,1},{64,64,1},
  {64,128,2},{128,128,1},{128,128,1},{128,128,1},
  {128,256,2},
  {256,256,1},{256,256,1},{256,256,1},{256,256,1},{256,256,1},{256,256,1},
  {256,256,1},{256,256,1},{256,256,1},{256,256,1},{256,256,1},{256,256,1},{256,256,1}
};

extern "C" void kernel_launch(void* const* d_in, const int* in_sizes, int n_in,
                              void* d_out, int out_size, void* d_ws, size_t ws_size,
                              hipStream_t stream)
{
  (void)in_sizes; (void)n_in; (void)out_size; (void)ws_size;

  // ---- gather input pointers in setup_inputs() dict order ----
  int ii = 0;
  auto F = [&]() { return (const float*)d_in[ii++]; };

  const float* x_in = F();
  const float* w_in = F();
  BNp bn_in = {F(), F(), F(), F()};
  const float* pr_in = F();

  Block blk[21];
  for (int i = 0; i < 21; ++i) {
    blk[i].bn1   = {F(), F(), F(), F()};
    blk[i].conv1 = F();
    blk[i].prelu = F();
    blk[i].conv2 = F();
    blk[i].bn2   = {F(), F(), F(), F()};
    blk[i].has_sc = (CFG[i][0] != CFG[i][1]);
    if (blk[i].has_sc) { blk[i].scw = F(); blk[i].scbn = {F(), F(), F(), F()}; }
    else               { blk[i].scw = nullptr; blk[i].scbn = {0,0,0,0}; }
  }
  const float *peW[3], *peB[3];
  for (int i = 0; i < 3; ++i) { peW[i] = F(); peB[i] = F(); }   // pe5, pe4, pe3
  Layer lay[3][6];
  for (int s = 0; s < 3; ++s)                                   // layers5,4,3
    for (int l = 0; l < 6; ++l) {
      lay[s][l].norm = F(); lay[s][l].inw = F(); lay[s][l].outw = F();
      for (int dd = 0; dd < 2; ++dd) {
        DirP& p = lay[s][l].dir[dd];
        p.cw = F(); p.cb = F(); p.xw = F(); p.dw = F(); p.db = F();
        p.alog = F(); p.D = F();
      }
    }
  Head hd[3];
  for (int s = 0; s < 3; ++s) { hd[s].bn = {F(), F(), F(), F()}; hd[s].lw = F(); hd[s].lb = F(); }

  // ---- workspace bump allocator ----
  char* wp = (char*)d_ws;
  auto alloc = [&](size_t bytes) {
    void* p = wp;
    wp += (bytes + 255) & ~(size_t)255;
    return p;
  };
  float*     A32  = (float*)alloc((size_t)16 * 256 * 256 * 64 * 4);
  float*     B32  = (float*)alloc((size_t)16 * 256 * 256 * 64 * 4);
  _Float16*  F16A = (_Float16*)alloc((size_t)16 * 256 * 256 * 64 * 2);
  _Float16*  F16B = (_Float16*)alloc((size_t)16 * 256 * 256 * 64 * 2);
  float*     SCB  = (float*)alloc((size_t)16 * 64 * 64 * 128 * 4);
  _Float16*  WBUF = (_Float16*)alloc((size_t)4096 * 4096 * 2);
  float*     S0   = (float*)alloc(4096 * 4);
  float*     S1   = (float*)alloc(4096 * 4);
  _Float16*  HN16 = (_Float16*)alloc((size_t)1024 * 512 * 2);
  float*     XZ   = (float*)alloc((size_t)1024 * 2048 * 4);
  float*     XL32 = (float*)alloc((size_t)1024 * 1024 * 4);
  _Float16*  XL16 = (_Float16*)alloc((size_t)1024 * 1024 * 2);
  float*     XD32 = (float*)alloc((size_t)1024 * 64 * 4);
  _Float16*  XD16 = (_Float16*)alloc((size_t)1024 * 64 * 2);
  float*     DTP  = (float*)alloc((size_t)1024 * 1024 * 4);
  float*     YC   = (float*)alloc((size_t)1024 * 1024 * 4);
  _Float16*  Y16  = (_Float16*)alloc((size_t)1024 * 1024 * 2);
  float*     X1A  = (float*)alloc((size_t)1024 * 512 * 4);
  float*     X1B  = (float*)alloc((size_t)1024 * 512 * 4);
  float*     RESA = (float*)alloc((size_t)1024 * 512 * 4);
  float*     RESB = (float*)alloc((size_t)1024 * 512 * 4);
  _Float16*  FEAT = (_Float16*)alloc((size_t)16 * 4096 * 2);

  // ---- launch helpers ----
  auto bnfold = [&](const BNp& p, int C) {
    k_bnfold<<<dim3((C + 255) / 256), 256, 0, stream>>>(p.g, p.b, p.m, p.v, S0, S1, C);
  };
  auto cvt16 = [&](const float* src, _Float16* dst, long long n, int C,
                   const float* s, const float* b) {
    k_affine16<<<dim3((unsigned)((n + 255) / 256)), 256, 0, stream>>>(src, dst, n, C, s, b);
  };
  auto repack = [&](const float* w, int O, int I, int R, int S) {
    long long n = (long long)O * I * R * S;
    k_repack<<<dim3((unsigned)((n + 255) / 256)), 256, 0, stream>>>(w, WBUF, O, I, R, S);
  };
  auto convL = [&](const _Float16* In, float* o32, _Float16* o16,
                   int Bn, int Hin, int Win, int Cin, int ldin,
                   int Hout, int Wout, int Cout, int ldout, int K,
                   int R, int S, int pad, int st,
                   const float* sc, const float* bi, const float* al,
                   const float* res, int resMode, int ldres) {
    int M = Bn * Hout * Wout;
    dim3 grid((M + 63) / 64, (Cout + 127) / 128);
    k_conv_wmma<<<grid, 256, 0, stream>>>(In, WBUF, o32, o16,
        Bn, Hin, Win, Cin, ldin, Hout, Wout, Cout, ldout, K, R, S, pad, st,
        sc, bi, al, res, resMode, ldres);
  };

  // ---- input stem ----
  bnfold(bn_in, 64);
  {
    long long total = (long long)16 * 256 * 256 * 64;
    k_input_conv<<<dim3((unsigned)((total + 255) / 256)), 256, 0, stream>>>(
        x_in, w_in, S0, S1, pr_in, A32);
  }
  float* cur = A32;
  float* nxt = B32;
  int H = 256, W = 256;
  int bi = 0;

  auto runBlocks = [&](int count) {
    for (int j = 0; j < count; ++j, ++bi) {
      const Block& bk = blk[bi];
      const int cin = CFG[bi][0], cout = CFG[bi][1], st = CFG[bi][2];
      const int Ho = H / st, Wo = W / st;
      if (bk.has_sc) {
        // shortcut: BN(conv1x1(x, stride))
        cvt16(cur, F16A, (long long)16 * H * W * cin, cin, nullptr, nullptr);
        repack(bk.scw, cout, cin, 1, 1);
        bnfold(bk.scbn, cout);
        convL(F16A, SCB, nullptr, 16, H, W, cin, cin, Ho, Wo, cout, cout,
              cin, 1, 1, 0, st, S0, S1, nullptr, nullptr, 0, 0);
      }
      // r = BN1(x) -> f16
      bnfold(bk.bn1, cin);
      cvt16(cur, F16A, (long long)16 * H * W * cin, cin, S0, S1);
      // r = PReLU(conv3x3(r))
      repack(bk.conv1, cout, cin, 3, 3);
      convL(F16A, nullptr, F16B, 16, H, W, cin, cin, H, W, cout, cout,
            9 * cin, 3, 3, 1, 1, nullptr, nullptr, bk.prelu, nullptr, 0, 0);
      // out = BN2(conv3x3(r, stride)) + shortcut
      repack(bk.conv2, cout, cout, 3, 3);
      bnfold(bk.bn2, cout);
      if (bk.has_sc)
        convL(F16B, nxt, nullptr, 16, H, W, cout, cout, Ho, Wo, cout, cout,
              9 * cout, 3, 3, 1, st, S0, S1, nullptr, SCB, 1, cout);
      else
        convL(F16B, nxt, nullptr, 16, H, W, cout, cout, Ho, Wo, cout, cout,
              9 * cout, 3, 3, 1, st, S0, S1, nullptr, cur, 2, cout);
      float* t = cur; cur = nxt; nxt = t;
      H = Ho; W = Wo;
    }
  };

  auto mambaStage = [&](int si, int P, int C, float* x1old) -> float* {
    // patch embed as PxP/stride-P conv -> x1 [1024 tokens][512] (+ prior x1)
    cvt16(cur, F16A, (long long)16 * H * W * C, C, nullptr, nullptr);
    repack(peW[si], 512, C, P, P);
    convL(F16A, X1A, nullptr, 16, H, W, C, C, H / P, W / P, 512, 512,
          P * P * C, P, P, 0, P, nullptr, peB[si], nullptr,
          x1old, x1old ? 1 : 0, 512);
    float* x1 = X1A; float* x1n = X1B;
    float* res = nullptr; float* resN = RESA;
    for (int l = 0; l < 6; ++l) {
      const Layer& L = lay[si][l];
      k_rmsnorm_res<<<dim3(1024), 256, 0, stream>>>(x1, res, resN, HN16, L.norm);
      res = resN; resN = (res == RESA) ? RESB : RESA;
      // xz = hn @ in_proj.T  [1024][2048]
      repack(L.inw, 2048, 512, 1, 1);
      convL(HN16, XZ, nullptr, 1, 1, 1024, 512, 512, 1, 1024, 2048, 2048,
            512, 1, 1, 0, 1, nullptr, nullptr, nullptr, nullptr, 0, 0);
      for (int dd = 0; dd < 2; ++dd) {
        const DirP& p = L.dir[dd];
        k_conv1d_silu<<<dim3((16 * 64 * 1024 + 255) / 256), 256, 0, stream>>>(
            XZ, p.cw, p.cb, XL32, XL16, dd);
        // x_dbl = xl @ x_proj.T  [1024][64]
        repack(p.xw, 64, 1024, 1, 1);
        convL(XL16, XD32, nullptr, 1, 1, 1024, 1024, 1024, 1, 1024, 64, 64,
              1024, 1, 1, 0, 1, nullptr, nullptr, nullptr, nullptr, 0, 0);
        cvt16(XD32, XD16, (long long)1024 * 64, 64, nullptr, nullptr);
        // delta_pre = dt @ dt_proj.T  [1024][1024]  (dt = cols 0..31, ldin=64)
        repack(p.dw, 1024, 32, 1, 1);
        convL(XD16, DTP, nullptr, 1, 1, 1024, 32, 64, 1, 1024, 1024, 1024,
              32, 1, 1, 0, 1, nullptr, nullptr, nullptr, nullptr, 0, 0);
        k_scan<<<dim3((16 * 1024 + 255) / 256), 256, 0, stream>>>(
            DTP, p.db, p.alog, XD32, XL32, p.D, XZ, YC, dd, dd);
      }
      cvt16(YC, Y16, (long long)1024 * 1024, 1024, nullptr, nullptr);
      // x1' = y @ out_proj.T  [1024][512]
      repack(L.outw, 512, 1024, 1, 1);
      convL(Y16, x1n, nullptr, 1, 1, 1024, 1024, 1024, 1, 1024, 512, 512,
            1024, 1, 1, 0, 1, nullptr, nullptr, nullptr, nullptr, 0, 0);
      float* t = x1; x1 = x1n; x1n = t;
    }
    // head: BN over tokens, mean-pool by 8, linear -> slice of d_out
    const Head& hh = hd[si];
    bnfold(hh.bn, 64);
    k_head_pre<<<dim3((16 * 4096 + 255) / 256), 256, 0, stream>>>(x1, S0, S1, FEAT);
    const int kk  = (si == 2) ? 8 : 4;
    const int off = (si == 0) ? 12 : (si == 1) ? 8 : 0;
    repack(hh.lw, kk * 512, 4096, 1, 1);
    convL(FEAT, (float*)d_out + off * 512, nullptr, 1, 1, 16, 4096, 4096,
          1, 16, kk * 512, 16 * 512, 4096, 1, 1, 0, 1,
          nullptr, hh.lb, nullptr, nullptr, 0, 0);
    return x1;
  };

  // ---- full pipeline (matches reference control flow) ----
  runBlocks(3);                                   // body[0:3]   -> 128x128 x64
  float* x1 = mambaStage(0, 16, 64, nullptr);     // pe5 + layers5 + out5 (lc4)
  runBlocks(4);                                   // body[3:7]   -> 64x64 x128
  x1 = mambaStage(1, 8, 128, x1);                 // pe4 + layers4 + out4 (lc3)
  runBlocks(14);                                  // body[7:21]  -> 32x32 x256
  x1 = mambaStage(2, 4, 256, x1);                 // pe3 + layers3 + out3 (lc2)
  (void)x1;
}
